// CTRNNCell_44375602102289
// MI455X (gfx1250) — compile-verified, matrix operand order checked by reading, not run on hardware
//
#include <hip/hip_runtime.h>
#include <math.h>

#define NDIM 1024
#define BROWS 2048
#define MAX_STEPS 40

typedef _Float16 half8_t __attribute__((ext_vector_type(8)));
typedef _Float16 v16h    __attribute__((ext_vector_type(16)));
typedef float    v8f     __attribute__((ext_vector_type(8)));

// scal layout (floats): [0]=t [1]=dt [2]=h [3]=done [4]=step_ok [5]=sumsq

// ---------------------------------------------------------------- init
__global__ void ctrnn_init(const float* __restrict__ inputs,
                           const float* __restrict__ prev,
                           const float* __restrict__ tau,
                           const float* __restrict__ W,
                           const float* __restrict__ iw,
                           const float* __restrict__ bias,
                           float* __restrict__ y,
                           float* __restrict__ drive,
                           float* __restrict__ inv_tau,
                           _Float16* __restrict__ Wh,
                           _Float16* __restrict__ act1,
                           float* __restrict__ scal)
{
    size_t idx = (size_t)blockIdx.x * blockDim.x + threadIdx.x;
    const size_t BN = (size_t)BROWS * NDIM;
    const size_t NN = (size_t)NDIM * NDIM;
    if (idx < BN) {
        float yv = prev[idx];
        y[idx]     = yv;
        drive[idx] = inputs[idx] * iw[idx & (NDIM - 1)];
        act1[idx]  = (_Float16)tanhf(yv + bias[idx & (NDIM - 1)]);
    }
    if (idx < NN)   Wh[idx]      = (_Float16)W[idx];
    if (idx < NDIM) inv_tau[idx] = 1.0f / tau[idx];
    if (idx == 0) {
        scal[0] = 0.0f;   // t
        scal[1] = 0.1f;   // dt
        scal[2] = 0.1f;   // h for step 1 = min(dt, 1-t)
        scal[3] = 0.0f;   // done
        scal[4] = 0.0f;   // step_ok
        scal[5] = 0.0f;   // sumsq
    }
}

// ------------------------------------------------- per-step scalar ctrl
// Folds step-finalization AND next step's h/done computation.
__global__ void ctrnn_finish(float* __restrict__ scal)
{
    float t = scal[0], dt = scal[1], h = scal[2];
    bool  done  = scal[3] != 0.0f;
    float sumsq = scal[5];
    float enorm = sqrtf(sumsq / (float)((size_t)BROWS * NDIM));
    enorm = fmaxf(enorm, 1e-10f);
    bool accept  = enorm <= 1.0f;
    bool step_ok = accept && !done;
    float tn = step_ok ? (t + h) : t;
    float factor = 0.9f * powf(enorm, -0.2f);
    factor = fminf(fmaxf(factor, 0.2f), 5.0f);
    float dtn = done ? dt : dt * factor;
    scal[0] = tn;
    scal[1] = dtn;
    scal[4] = step_ok ? 1.0f : 0.0f;
    scal[5] = 0.0f;                      // reset accumulator
    bool done2 = tn >= (1.0f - 1e-7f);   // prepare next step
    scal[3] = done2 ? 1.0f : 0.0f;
    scal[2] = done2 ? 0.0f : fminf(dtn, 1.0f - tn);
}

// commit y and produce next step's stage-1 activations
__global__ __launch_bounds__(256)
void ctrnn_commit(const float* __restrict__ scal,
                  const float* __restrict__ ys,
                  const float* __restrict__ bias,
                  float* __restrict__ y,
                  _Float16* __restrict__ act1)
{
    size_t idx = (size_t)blockIdx.x * blockDim.x + threadIdx.x;
    float yv = y[idx];
    float yn = (scal[4] != 0.0f) ? ys[idx] : yv;
    y[idx]    = yn;
    act1[idx] = (_Float16)tanhf(yn + bias[idx & (NDIM - 1)]);
}

// ------------------------------------------------- WMMA tile loader
__device__ __forceinline__ v16h load_tile16(const _Float16* p)
{
    half8_t lo = *(const half8_t*)(p);
    half8_t hi = *(const half8_t*)(p + 16);
    return __builtin_shufflevector(lo, hi, 0,1,2,3,4,5,6,7,8,9,10,11,12,13,14,15);
}

// ------------------------------------------------- fused GEMM stage S
// Computes k_S = inv_tau*(Act_S@W^T + drive - ys_S) and, in the same
// epilogue, the next stage input: ys_{S+1} = y + h*sum(a_{S+1,j} k_j)
// and act_{S+1} = tanh(ys_{S+1} + bias).
// Per wave: 32(M) x 64(N) tile -> 2 A tiles x 4 B tiles = 8 WMMAs/K-step.
template <int S>
__global__ __launch_bounds__(128)
void ctrnn_gemm_fused(const _Float16* __restrict__ actin,
                      const _Float16* __restrict__ Wh,
                      const float* __restrict__ ysin,
                      const float* __restrict__ drive,
                      const float* __restrict__ invt,
                      const float* __restrict__ y,
                      const float* __restrict__ bias,
                      const float* __restrict__ scal,
                      const float* __restrict__ p1, const float* __restrict__ p2,
                      const float* __restrict__ p3, const float* __restrict__ p4,
                      const float* __restrict__ p5,
                      float* __restrict__ kout,
                      float* __restrict__ ysout,
                      _Float16* __restrict__ actout)
{
    const int N = NDIM;
    int lane = threadIdx.x & 31;
    int wave = threadIdx.x >> 5;
    int r  = lane & 15;
    int cs = lane >> 4;
    int b0 = blockIdx.x * 128 + wave * 32;
    int n0 = blockIdx.y * 64;

    const _Float16* pA0 = actin + (size_t)(b0 + r) * N + cs * 8;
    const _Float16* pA1 = pA0 + (size_t)16 * N;
    const _Float16* pB  = Wh + (size_t)(n0 + r) * N + cs * 8;

    v8f acc[2][4] = {};
    for (int k0 = 0; k0 < N; k0 += 32) {
        v16h a0 = load_tile16(pA0 + k0);
        v16h a1 = load_tile16(pA1 + k0);
#pragma unroll
        for (int t = 0; t < 4; ++t) {
            v16h b = load_tile16(pB + (size_t)t * 16 * N + k0);
            acc[0][t] = __builtin_amdgcn_wmma_f32_16x16x32_f16(
                false, a0, false, b, (short)0, acc[0][t], false, false);
            acc[1][t] = __builtin_amdgcn_wmma_f32_16x16x32_f16(
                false, a1, false, b, (short)0, acc[1][t], false, false);
        }
    }

    float h = scal[2];
#pragma unroll
    for (int mt = 0; mt < 2; ++mt) {
#pragma unroll
        for (int t = 0; t < 4; ++t) {
            int col = n0 + t * 16 + r;
            float it = invt[col];
            float bb = bias[col];
#pragma unroll
            for (int v = 0; v < 8; ++v) {
                int row = b0 + mt * 16 + v + 8 * cs;
                size_t idx = (size_t)row * N + col;
                float kval = it * (acc[mt][t][v] + drive[idx] - ysin[idx]);
                kout[idx] = kval;
                float comb;
                if constexpr (S == 1)
                    comb = 0.2f * kval;
                else if constexpr (S == 2)
                    comb = (3.0f/40.0f) * p1[idx] + (9.0f/40.0f) * kval;
                else if constexpr (S == 3)
                    comb = (44.0f/45.0f) * p1[idx] + (-56.0f/15.0f) * p2[idx]
                         + (32.0f/9.0f) * kval;
                else if constexpr (S == 4)
                    comb = (19372.0f/6561.0f) * p1[idx] + (-25360.0f/2187.0f) * p2[idx]
                         + (64448.0f/6561.0f) * p3[idx] + (-212.0f/729.0f) * kval;
                else if constexpr (S == 5)
                    comb = (9017.0f/3168.0f) * p1[idx] + (-355.0f/33.0f) * p2[idx]
                         + (46732.0f/5247.0f) * p3[idx] + (49.0f/176.0f) * p4[idx]
                         + (-5103.0f/18656.0f) * kval;
                else // S == 6 -> 5th-order solution y5 (k2 coeff is 0)
                    comb = (35.0f/384.0f) * p1[idx] + (500.0f/1113.0f) * p3[idx]
                         + (125.0f/192.0f) * p4[idx] + (-2187.0f/6784.0f) * p5[idx]
                         + (11.0f/84.0f) * kval;
                float ynext = fmaf(h, comb, y[idx]);
                ysout[idx] = ynext;
                actout[idx] = (_Float16)tanhf(ynext + bb);
            }
        }
    }
}

// ------------------------------------------------- GEMM stage 7 (FSAL):
// k7 on the fly; accumulate sum((h*dot(E,k)/scale)^2) into scal[5]
__global__ __launch_bounds__(128)
void ctrnn_gemm_err(const _Float16* __restrict__ act,
                    const _Float16* __restrict__ Wh,
                    const float* __restrict__ y5,     // = ys buffer
                    const float* __restrict__ drive,
                    const float* __restrict__ invt,
                    const float* __restrict__ y,
                    const float* __restrict__ k1,
                    const float* __restrict__ k3,
                    const float* __restrict__ k4,
                    const float* __restrict__ k5,
                    const float* __restrict__ k6,
                    float* __restrict__ scal)
{
    constexpr float E1 =  71.0f / 57600.0f;
    constexpr float E3 = -71.0f / 16695.0f;
    constexpr float E4 =  71.0f / 1920.0f;
    constexpr float E5 = -17253.0f / 339200.0f;
    constexpr float E6 =  22.0f / 525.0f;
    constexpr float E7 = -1.0f / 40.0f;

    const int N = NDIM;
    int lane = threadIdx.x & 31;
    int wave = threadIdx.x >> 5;
    int r  = lane & 15;
    int cs = lane >> 4;
    int b0 = blockIdx.x * 128 + wave * 32;
    int n0 = blockIdx.y * 64;

    const _Float16* pA0 = act + (size_t)(b0 + r) * N + cs * 8;
    const _Float16* pA1 = pA0 + (size_t)16 * N;
    const _Float16* pB  = Wh + (size_t)(n0 + r) * N + cs * 8;

    v8f acc[2][4] = {};
    for (int k0 = 0; k0 < N; k0 += 32) {
        v16h a0 = load_tile16(pA0 + k0);
        v16h a1 = load_tile16(pA1 + k0);
#pragma unroll
        for (int t = 0; t < 4; ++t) {
            v16h b = load_tile16(pB + (size_t)t * 16 * N + k0);
            acc[0][t] = __builtin_amdgcn_wmma_f32_16x16x32_f16(
                false, a0, false, b, (short)0, acc[0][t], false, false);
            acc[1][t] = __builtin_amdgcn_wmma_f32_16x16x32_f16(
                false, a1, false, b, (short)0, acc[1][t], false, false);
        }
    }

    float h = scal[2];
    float lsum = 0.0f;
#pragma unroll
    for (int mt = 0; mt < 2; ++mt) {
#pragma unroll
        for (int t = 0; t < 4; ++t) {
            int col = n0 + t * 16 + r;
            float it = invt[col];
#pragma unroll
            for (int v = 0; v < 8; ++v) {
                int row = b0 + mt * 16 + v + 8 * cs;
                size_t idx = (size_t)row * N + col;
                float yv  = y[idx];
                float y5v = y5[idx];
                float k7  = it * (acc[mt][t][v] + drive[idx] - y5v);
                float e = h * (E1 * k1[idx] + E3 * k3[idx] + E4 * k4[idx] +
                               E5 * k5[idx] + E6 * k6[idx] + E7 * k7);
                float sc = 1e-6f + 1e-3f * fmaxf(fabsf(yv), fabsf(y5v));
                float q = e / sc;
                lsum += q * q;
            }
        }
    }

    __shared__ float red[128];
    red[threadIdx.x] = lsum;
    __syncthreads();
    for (int s = 64; s > 0; s >>= 1) {
        if ((int)threadIdx.x < s) red[threadIdx.x] += red[threadIdx.x + s];
        __syncthreads();
    }
    if (threadIdx.x == 0) atomicAdd(&scal[5], red[0]);
}

// ---------------------------------------------------------------- host
extern "C" void kernel_launch(void* const* d_in, const int* in_sizes, int n_in,
                              void* d_out, int out_size, void* d_ws, size_t ws_size,
                              hipStream_t stream)
{
    (void)in_sizes; (void)n_in; (void)out_size; (void)ws_size;
    const int    N  = NDIM, B = BROWS;
    const size_t BN = (size_t)B * N;
    const size_t NN = (size_t)N * N;

    const float* inputs = (const float*)d_in[0];
    const float* prev   = (const float*)d_in[1];
    const float* tau    = (const float*)d_in[2];
    const float* W      = (const float*)d_in[3];
    const float* iw     = (const float*)d_in[4];
    const float* bias   = (const float*)d_in[5];
    float* y = (float*)d_out;     // live state, committed in place

    char* ws = (char*)d_ws;
    size_t off = 0;
    float* scal = (float*)(ws + off); off += 256;
    float* kb[6];
    for (int i = 0; i < 6; ++i) { kb[i] = (float*)(ws + off); off += BN * sizeof(float); }
    float* ys    = (float*)(ws + off); off += BN * sizeof(float);   // doubles as y5
    float* drive = (float*)(ws + off); off += BN * sizeof(float);
    float* invt  = (float*)(ws + off); off += 4096;
    _Float16* Wh   = (_Float16*)(ws + off); off += NN * sizeof(_Float16);
    _Float16* actA = (_Float16*)(ws + off); off += BN * sizeof(_Float16); // ping
    _Float16* actB = (_Float16*)(ws + off); off += BN * sizeof(_Float16); // pong

    dim3 eb(256), eg((unsigned)((BN + 255) / 256));
    dim3 gb(128), gg(B / 128, N / 64);

    ctrnn_init<<<eg, eb, 0, stream>>>(inputs, prev, tau, W, iw, bias,
                                      y, drive, invt, Wh, actA, scal);

    for (int step = 0; step < MAX_STEPS; ++step) {
        // stage 1: k1 = f(y); fused: build stage-2 input (act ping->pong)
        ctrnn_gemm_fused<1><<<gg, gb, 0, stream>>>(actA, Wh, y, drive, invt,
            y, bias, scal, nullptr, nullptr, nullptr, nullptr, nullptr,
            kb[0], ys, actB);
        // stage 2 (pong->ping)
        ctrnn_gemm_fused<2><<<gg, gb, 0, stream>>>(actB, Wh, ys, drive, invt,
            y, bias, scal, kb[0], nullptr, nullptr, nullptr, nullptr,
            kb[1], ys, actA);
        // stage 3 (ping->pong)
        ctrnn_gemm_fused<3><<<gg, gb, 0, stream>>>(actA, Wh, ys, drive, invt,
            y, bias, scal, kb[0], kb[1], nullptr, nullptr, nullptr,
            kb[2], ys, actB);
        // stage 4 (pong->ping)
        ctrnn_gemm_fused<4><<<gg, gb, 0, stream>>>(actB, Wh, ys, drive, invt,
            y, bias, scal, kb[0], kb[1], kb[2], nullptr, nullptr,
            kb[3], ys, actA);
        // stage 5 (ping->pong)
        ctrnn_gemm_fused<5><<<gg, gb, 0, stream>>>(actA, Wh, ys, drive, invt,
            y, bias, scal, kb[0], kb[1], kb[2], kb[3], nullptr,
            kb[4], ys, actB);
        // stage 6 (pong->ping): also produces y5 into ys and act7 into actA
        ctrnn_gemm_fused<6><<<gg, gb, 0, stream>>>(actB, Wh, ys, drive, invt,
            y, bias, scal, kb[0], kb[1], kb[2], kb[3], kb[4],
            kb[5], ys, actA);
        // stage 7: fused k7 + error-norm accumulation
        ctrnn_gemm_err<<<gg, gb, 0, stream>>>(actA, Wh, ys, drive, invt, y,
            kb[0], kb[2], kb[3], kb[4], kb[5], scal);

        // finalize step + compute next step's h/done
        ctrnn_finish<<<1, 1, 0, stream>>>(scal);
        // commit y and next step's stage-1 activations (into ping buffer)
        ctrnn_commit<<<eg, eb, 0, stream>>>(scal, ys, bias, y, actA);
    }
}
